// DeepME_Model_30674656428508
// MI455X (gfx1250) — matrix operand order repaired and checked by reading, not compiled, not run on hardware
//
#include <hip/hip_runtime.h>
#include <hip/hip_bf16.h>

#define H    192
#define H3   64
#define EPP  64               // edges per wave-pair (4 M-subtiles of 16)
#define PAIRS 2               // pairs per block -> 128 threads, 4 waves, fills WGP
#define BLOCK_EDGES (EPP * PAIRS)

typedef _Float16 v16h __attribute__((ext_vector_type(16)));
typedef _Float16 v8h  __attribute__((ext_vector_type(8)));
typedef _Float16 v4h  __attribute__((ext_vector_type(4)));
typedef float    v8f  __attribute__((ext_vector_type(8)));

// ---- packed-weight layout in d_ws (units: _Float16 elements) -------------
// fragment = 32(K) x 16(N) f16, lane-major: frag*512 + lane*16 + elem
// frag order: nTile*kTiles + kTile
#define OFF_WS   0         // 192x192 : 6x12 = 72 frags
#define OFF_WD   36864
#define OFF_W1   73728     // 192x64  : 6x4  = 24 frags
#define OFF_W2   86016
#define OFF_W3   98304
#define OFF_WF1  110592    // 576x192 : 18x12 = 216 frags
#define OFF_WF2  221184    // 192x192 : 72 frags
#define OFF_WF3  258048    // 192x3 (padded N->16) : 6 frags

// ---- per-pair LDS slice --------------------------------------------------
#define SM_SRC    0        // 64x192 f16 = 24576  (reused as hBuf)
#define SM_DST    24576    // 64x192 f16 = 24576  (reused as h2Buf)
#define SM_CCAT   49152    // 64x576 f16 = 73728  (reused as 64x16 f32 logits)
#define SM_SIDX   122880   // 64 int
#define SM_DIDX   123136   // 64 int
#define PAIR_SMEM 123392   // x2 pairs = 246784 B of 320 KB WGP LDS

// ===========================================================================
// Weight pack: W[K][N] f32 row-major -> WMMA B-fragment layout (f16).
// B 32x16 f16: lanes 0-15 hold col n=lane K=0..15; lanes 16-31 col n, K=16..31
// ===========================================================================
__global__ void pack_weight(const float* __restrict__ W, _Float16* __restrict__ dst,
                            int K, int N, int kTiles, int nTiles) {
    int idx = blockIdx.x * blockDim.x + threadIdx.x;
    int total = kTiles * nTiles * 512;
    if (idx >= total) return;
    int frag = idx >> 9;
    int rem  = idx & 511;
    int l    = rem >> 4;
    int i    = rem & 15;
    int nTile = frag / kTiles;
    int kTile = frag - nTile * kTiles;
    int n = nTile * 16 + (l & 15);
    int k = kTile * 32 + (l >> 4) * 16 + i;
    float v = (k < K && n < N) ? W[k * N + n] : 0.f;
    dst[idx] = (_Float16)v;
}

// ===========================================================================
// A-fragment from LDS row-major f16 [16][stride]:
// lane l<16 : row l,    elems0-15 = K0..15
// lane l>=16: row l-16, elems0-15 = K16..31
// ===========================================================================
__device__ inline v16h load_a(const _Float16* buf, int stride, int lane, int k0) {
    int row  = lane & 15;
    int half = lane >> 4;
    const _Float16* p = buf + row * stride + k0 + half * 16;
    v8h lo = *(const v8h*)p;
    v8h hi = *(const v8h*)(p + 8);
    v16h a;
#pragma unroll
    for (int i = 0; i < 8; ++i) { a[i] = lo[i]; a[i + 8] = hi[i]; }
    return a;
}

__device__ inline v16h load_b(const _Float16* pk, int frag, int lane) {
    return *(const v16h*)(pk + (size_t)frag * 512 + (size_t)lane * 16); // 32B coalesced
}

// MODE: 0=src, 1=dst, 2=src-dst, 3=(src-dst)^2, 4=src*dst (packed f16 VALU,
// co-issues next to XDL WMMA). With nt-pair blocking each derived fragment
// feeds TWO WMMAs, halving this VALU work and the DS A-traffic.
template <int MODE>
__device__ inline v16h load_a_mode(const _Float16* s, const _Float16* d, int lane, int k0) {
    v16h a = load_a(s, H, lane, k0);
    if (MODE == 0) return a;
    v16h b = load_a(d, H, lane, k0);
    if (MODE == 1) return b;
    v16h r;
#pragma unroll
    for (int i = 0; i < 16; ++i) {
        if (MODE == 2)      r[i] = a[i] - b[i];
        else if (MODE == 3) { _Float16 t = (_Float16)(a[i] - b[i]); r[i] = (_Float16)(t * t); }
        else                r[i] = a[i] * b[i];
    }
    return r;
}

// shared epilogue: bias + ReLU, D-layout scatter to f16 LDS
__device__ inline void store_relu(const v8f* acc, const float* __restrict__ bias,
                                  _Float16* dst, int dstStride, int dstOff,
                                  int nt, int lane) {
    int n  = nt * 16 + (lane & 15);
    float bv = bias[n];
    int mb = (lane >> 4) * 8;
#pragma unroll
    for (int m = 0; m < 4; ++m)
#pragma unroll
        for (int r = 0; r < 8; ++r) {
            float v = acc[m][r] + bv;
            dst[(m * 16 + mb + r) * dstStride + dstOff + n] = (_Float16)(v > 0.f ? v : 0.f);
        }
}

// ---------------------------------------------------------------------------
// Branch GEMM: M-blocked x4 AND N-pair blocked x2 -> each A fragment (one DS
// read + one MODE compute) feeds 2 WMMAs; each B fragment feeds 4 WMMAs.
// Waves of a pair split n-tiles; pairs are (w2, w2+2), nT in {4,12}.
// ---------------------------------------------------------------------------
template <int MODE>
__device__ inline void branch_gemm(const _Float16* s, const _Float16* d,
                                   const _Float16* pk, const float* __restrict__ bias,
                                   _Float16* dst, int dstStride, int dstOff,
                                   int Nout, int lane, int w2) {
    int nT = Nout >> 4;
    for (int nt = w2; nt < nT; nt += 4) {
        int ntB = nt + 2;
        v8f acc0[4] = {}, acc1[4] = {};
#pragma unroll 2
        for (int kt = 0; kt < 6; ++kt) {               // K = 192
            v16h b0 = load_b(pk, nt  * 6 + kt, lane);
            v16h b1 = load_b(pk, ntB * 6 + kt, lane);
#pragma unroll
            for (int m = 0; m < 4; ++m) {
                v16h a = load_a_mode<MODE>(s + m * 16 * H, d + m * 16 * H, lane, kt * 32);
                acc0[m] = __builtin_amdgcn_wmma_f32_16x16x32_f16(false, a, false, b0,
                                                                 (short)0, acc0[m], false, false);
                acc1[m] = __builtin_amdgcn_wmma_f32_16x16x32_f16(false, a, false, b1,
                                                                 (short)0, acc1[m], false, false);
            }
        }
        store_relu(acc0, bias, dst, dstStride, dstOff, nt,  lane);
        store_relu(acc1, bias, dst, dstStride, dstOff, ntB, lane);
    }
}

// generic M-blocked + N-pair blocked GEMM, f16 A [64][Kin] -> ReLU f16 [64][Nout]
__device__ inline void gemm_f16out(const _Float16* A, int Kin,
                                   const _Float16* pk, const float* __restrict__ bias,
                                   _Float16* dst, int Nout, int lane, int w2) {
    int kT = Kin >> 5, nT = Nout >> 4;
    for (int nt = w2; nt < nT; nt += 4) {
        int ntB = nt + 2;
        v8f acc0[4] = {}, acc1[4] = {};
#pragma unroll 2
        for (int kt = 0; kt < kT; ++kt) {
            v16h b0 = load_b(pk, nt  * kT + kt, lane);
            v16h b1 = load_b(pk, ntB * kT + kt, lane);
#pragma unroll
            for (int m = 0; m < 4; ++m) {
                v16h a = load_a(A + m * 16 * Kin, Kin, lane, kt * 32);
                acc0[m] = __builtin_amdgcn_wmma_f32_16x16x32_f16(false, a, false, b0,
                                                                 (short)0, acc0[m], false, false);
                acc1[m] = __builtin_amdgcn_wmma_f32_16x16x32_f16(false, a, false, b1,
                                                                 (short)0, acc1[m], false, false);
            }
        }
        store_relu(acc0, bias, dst, Nout, 0, nt,  lane);
        store_relu(acc1, bias, dst, Nout, 0, ntB, lane);
    }
}

// final 192 -> 16 (cols 0..2 valid), raw f32 logits [64][16]
__device__ inline void gemm_raw16(const _Float16* A, const _Float16* pk,
                                  float* out, int lane) {
    v8f acc[4] = {};
#pragma unroll 2
    for (int kt = 0; kt < 6; ++kt) {
        v16h b = load_b(pk, kt, lane);
#pragma unroll
        for (int m = 0; m < 4; ++m) {
            v16h a = load_a(A + m * 16 * H, H, lane, kt * 32);
            acc[m] = __builtin_amdgcn_wmma_f32_16x16x32_f16(false, a, false, b,
                                                            (short)0, acc[m], false, false);
        }
    }
    int n  = lane & 15;
    int mb = (lane >> 4) * 8;
#pragma unroll
    for (int m = 0; m < 4; ++m)
#pragma unroll
        for (int r = 0; r < 8; ++r) out[(m * 16 + mb + r) * 16 + n] = acc[m][r];
}

// In-place row-private LayerNorm: thread ptid owns row ptid (64 rows <-> 64
// threads per pair). Stats stay in registers; f32 accumulate over f16 data.
__device__ inline void ln_rows(_Float16* buf, int stride, int off, int Nout,
                               const float* __restrict__ g, const float* __restrict__ be,
                               int ptid) {
    _Float16* row = buf + ptid * stride + off;
    float s = 0.f;
    for (int n = 0; n < Nout; n += 8) {
        v8h v = *(const v8h*)(row + n);
#pragma unroll
        for (int i = 0; i < 8; ++i) s += (float)v[i];
    }
    float mu = s / (float)Nout;
    float vs = 0.f;
    for (int n = 0; n < Nout; n += 8) {
        v8h v = *(const v8h*)(row + n);
#pragma unroll
        for (int i = 0; i < 8; ++i) { float a = (float)v[i] - mu; vs += a * a; }
    }
    float rs = rsqrtf(vs / (float)Nout + 1e-5f);
    for (int n = 0; n < Nout; n += 8) {
        v8h v = *(const v8h*)(row + n);
        v8h o;
#pragma unroll
        for (int i = 0; i < 8; ++i)
            o[i] = (_Float16)(((float)v[i] - mu) * rs * g[n + i] + be[n + i]);
        *(v8h*)(row + n) = o;
    }
}

// ===========================================================================
// Main kernel: 2 wave-pairs per block; each pair owns a 64-edge tile.
// Waves in a pair split n-tiles; pipeline is uniform -> plain __syncthreads.
// ===========================================================================
__global__ void __launch_bounds__(PAIRS * 64)
edge_mlp(const float* __restrict__ kge, const int* __restrict__ eidx,
         const int* __restrict__ etype, int E,
         const _Float16* __restrict__ pack,
         const float* b1, const float* g1, const float* be1,
         const float* b2, const float* g2, const float* be2,
         const float* b3, const float* g3, const float* be3,
         const float* bs, const float* gs, const float* bes,
         const float* bd, const float* gd, const float* bed,
         const float* bf1, const float* gf, const float* bef,
         const float* bf2, const float* bf3,
         float* __restrict__ out) {
    extern __shared__ char smemRaw[];
    const int tid  = threadIdx.x;
    const int lane = tid & 31;
    const int w2   = (tid >> 5) & 1;   // wave within pair (n-tile split)
    const int pair = tid >> 6;
    const int ptid = tid & 63;         // thread within pair

    char* base = smemRaw + (size_t)pair * PAIR_SMEM;
    _Float16* srcF  = (_Float16*)(base + SM_SRC);
    _Float16* dstF  = (_Float16*)(base + SM_DST);
    _Float16* ccat  = (_Float16*)(base + SM_CCAT);
    int*      sIdx  = (int*)(base + SM_SIDX);
    int*      dIdx  = (int*)(base + SM_DIDX);
    _Float16* hBuf  = srcF;                 // fusion reuses src tile
    _Float16* h2Buf = dstF;                 // and dst tile
    float*    logit = (float*)ccat;         // 64x16 f32, reuses concat

    const int tbase = (blockIdx.x * PAIRS + pair) * EPP;

    // 1) edge endpoints (clamped; dummy rows never store)
    {
        int e = tbase + ptid;
        if (e >= E) e = E - 1;
        sIdx[ptid] = eidx[e];
        dIdx[ptid] = eidx[E + e];
    }
    __syncthreads();

    // 2) gather node rows (float4 from HBM) -> f16 LDS tiles [64][192]
    const float4* kge4 = (const float4*)kge;   // row = 48 float4
    for (int i = ptid; i < EPP * 48; i += 64) {
        int r = i / 48;
        int c = i - r * 48;
        float4 vs = kge4[(size_t)sIdx[r] * 48 + c];
        float4 vd = kge4[(size_t)dIdx[r] * 48 + c];
        v4h hs, hd;
        hs[0] = (_Float16)vs.x; hs[1] = (_Float16)vs.y;
        hs[2] = (_Float16)vs.z; hs[3] = (_Float16)vs.w;
        hd[0] = (_Float16)vd.x; hd[1] = (_Float16)vd.y;
        hd[2] = (_Float16)vd.z; hd[3] = (_Float16)vd.w;
        *(v4h*)(srcF + r * H + c * 4) = hs;
        *(v4h*)(dstF + r * H + c * 4) = hd;
    }
    __syncthreads();

    // 3) five branch GEMMs (pre-LN, bias+ReLU) -> ccat[64][576]
    branch_gemm<0>(srcF, dstF, pack + OFF_WS, bs, ccat, 3 * H, 0,            H,  lane, w2);
    branch_gemm<1>(srcF, dstF, pack + OFF_WD, bd, ccat, 3 * H, H,            H,  lane, w2);
    branch_gemm<2>(srcF, dstF, pack + OFF_W1, b1, ccat, 3 * H, 2 * H,        H3, lane, w2);
    branch_gemm<3>(srcF, dstF, pack + OFF_W2, b2, ccat, 3 * H, 2 * H + H3,   H3, lane, w2);
    branch_gemm<4>(srcF, dstF, pack + OFF_W3, b3, ccat, 3 * H, 2 * H + 2*H3, H3, lane, w2);
    __syncthreads();

    // 4) LayerNorm all five segments in place (row-private)
    ln_rows(ccat, 3 * H, 0,            H,  gs, bes, ptid);
    ln_rows(ccat, 3 * H, H,            H,  gd, bed, ptid);
    ln_rows(ccat, 3 * H, 2 * H,        H3, g1, be1, ptid);
    ln_rows(ccat, 3 * H, 2 * H + H3,   H3, g2, be2, ptid);
    ln_rows(ccat, 3 * H, 2 * H + 2*H3, H3, g3, be3, ptid);
    __syncthreads();

    // 5) fusion: 576 -> 192 (pre-LN) -> hBuf
    gemm_f16out(ccat, 3 * H, pack + OFF_WF1, bf1, hBuf, H, lane, w2);
    __syncthreads();
    ln_rows(hBuf, H, 0, H, gf, bef, ptid);
    __syncthreads();

    // 6) 192 -> 192 ReLU -> h2Buf
    gemm_f16out(hBuf, H, pack + OFF_WF2, bf2, h2Buf, H, lane, w2);
    __syncthreads();

    // 7) 192 -> 3 (padded 16): single n-tile, wave 0 of pair
    if (w2 == 0) gemm_raw16(h2Buf, pack + OFF_WF3, logit, lane);
    __syncthreads();

    // 8) softmax over 3 logits, select by edge_type
    {
        int e = tbase + ptid;
        if (e < E) {
            float l0 = logit[ptid * 16 + 0] + bf3[0];
            float l1 = logit[ptid * 16 + 1] + bf3[1];
            float l2 = logit[ptid * 16 + 2] + bf3[2];
            float mx = fmaxf(l0, fmaxf(l1, l2));
            float e0 = __expf(l0 - mx), e1 = __expf(l1 - mx), e2 = __expf(l2 - mx);
            float inv = 1.f / (e0 + e1 + e2);
            int t = etype[e];
            out[e] = (t == 0 ? e0 : (t == 1 ? e1 : e2)) * inv;
        }
    }
}

// ===========================================================================
extern "C" void kernel_launch(void* const* d_in, const int* in_sizes, int n_in,
                              void* d_out, int out_size, void* d_ws, size_t ws_size,
                              hipStream_t stream) {
    const float* kge   = (const float*)d_in[0];
    const int*   eidx  = (const int*)d_in[1];
    const int*   etype = (const int*)d_in[2];
    const float *W1 = (const float*)d_in[3],  *b1 = (const float*)d_in[4],
                *g1 = (const float*)d_in[5],  *be1 = (const float*)d_in[6];
    const float *W2 = (const float*)d_in[7],  *b2 = (const float*)d_in[8],
                *g2 = (const float*)d_in[9],  *be2 = (const float*)d_in[10];
    const float *W3 = (const float*)d_in[11], *b3 = (const float*)d_in[12],
                *g3 = (const float*)d_in[13], *be3 = (const float*)d_in[14];
    const float *Ws = (const float*)d_in[15], *bs = (const float*)d_in[16],
                *gs = (const float*)d_in[17], *bes = (const float*)d_in[18];
    const float *Wd = (const float*)d_in[19], *bd = (const float*)d_in[20],
                *gd = (const float*)d_in[21], *bed = (const float*)d_in[22];
    const float *Wf1 = (const float*)d_in[23], *bf1 = (const float*)d_in[24],
                *gf  = (const float*)d_in[25], *bef = (const float*)d_in[26];
    const float *Wf2 = (const float*)d_in[27], *bf2 = (const float*)d_in[28];
    const float *Wf3 = (const float*)d_in[29], *bf3 = (const float*)d_in[30];

    const int E = in_sizes[2];
    _Float16* pack = (_Float16*)d_ws;   // 522 KB packed weights, L2-resident

    auto pk = [&](const float* W, int off, int K, int N, int kT, int nT) {
        int total = kT * nT * 512;
        pack_weight<<<(total + 255) / 256, 256, 0, stream>>>(W, pack + off, K, N, kT, nT);
    };
    pk(Ws,  OFF_WS,  192, 192,  6, 12);
    pk(Wd,  OFF_WD,  192, 192,  6, 12);
    pk(W1,  OFF_W1,  192,  64,  6,  4);
    pk(W2,  OFF_W2,  192,  64,  6,  4);
    pk(W3,  OFF_W3,  192,  64,  6,  4);
    pk(Wf1, OFF_WF1, 576, 192, 18, 12);
    pk(Wf2, OFF_WF2, 192, 192,  6, 12);
    pk(Wf3, OFF_WF3, 192,   3,  6,  1);

    const int blocks = (E + BLOCK_EDGES - 1) / BLOCK_EDGES;
    const size_t shmem = (size_t)PAIR_SMEM * PAIRS;   // 246,784 B of 320 KB
    edge_mlp<<<blocks, PAIRS * 64, shmem, stream>>>(
        kge, eidx, etype, E, pack,
        b1, g1, be1, b2, g2, be2, b3, g3, be3,
        bs, gs, bes, bd, gd, bed,
        bf1, gf, bef, bf2, bf3,
        (float*)d_out);
}
